// Indexer_21199958573396
// MI455X (gfx1250) — matrix operand order repaired and checked by reading, not compiled, not run on hardware
//
#include <hip/hip_runtime.h>
#include <hip/hip_bf16.h>

#define BB   2
#define SS   2048
#define HH   2048
#define NHH  8
#define HDD  64
#define LATT 512
#define TOPKK 256

typedef __attribute__((ext_vector_type(16))) __bf16 v16bf;
typedef __attribute__((ext_vector_type(8)))  float  v8f;

// ---- WMMA helpers ------------------------------------------------------
__device__ __forceinline__ v8f bwmma(v16bf a, v16bf b, v8f c) {
    // D = A(16x32 bf16) * B(32x16 bf16) + C(16x16 f32)
    return __builtin_amdgcn_wmma_f32_16x16x32_bf16(
        false, a, false, b, (short)0, c, false, false);
}

// A fragment: two 16-byte chunks (K {k0..k0+7} and {k0+8..k0+15} per lane-half)
__device__ __forceinline__ v16bf load_frag2(const __bf16* p0, const __bf16* p1) {
    union { uint4 u[2]; v16bf v; } x;
    x.u[0] = *(const uint4*)p0;
    x.u[1] = *(const uint4*)p1;
    return x.v;
}
// B fragment: 16 contiguous bf16 (32 bytes)
__device__ __forceinline__ v16bf load_frag1(const __bf16* p) {
    union { uint4 u[2]; v16bf v; } x;
    x.u[0] = *(const uint4*)p;
    x.u[1] = *(const uint4*)(p + 8);
    return x.v;
}

__device__ __forceinline__ unsigned key_of(float f) {
    unsigned u = __float_as_uint(f);
    return (u & 0x80000000u) ? ~u : (u | 0x80000000u);   // order-preserving
}

// ---- kernel 0: convert weights to bf16 (runs once, tiny) ---------------
// wp_w is padded to 16 output rows (rows 8..15 zero) so the gate can run
// through the WMMA pipe with a full 16-column N-tile.
__global__ void idx_convert_w(const float* __restrict__ wq_b,
                              const float* __restrict__ wk_w,
                              const float* __restrict__ wp_w,
                              __bf16* __restrict__ wqb,
                              __bf16* __restrict__ wkb,
                              __bf16* __restrict__ wpb) {
    int i = blockIdx.x * blockDim.x + threadIdx.x;
    if (i < LATT * NHH * HDD) wqb[i] = (__bf16)wq_b[i];                     // 512*512
    if (i < HDD * HH)         wkb[i] = (__bf16)wk_w[i];                     // 64*2048
    if (i < 16 * HH)          wpb[i] = (i < NHH * HH) ? (__bf16)wp_w[i]
                                                      : (__bf16)0.f;       // 16*2048, padded
}

// ---- kernel 1: per 16-row tile: k = rope(LN(x Wk^T)), qeff = rope(sum_h g_h q_h)*scale
__global__ void idx_prep(const float* __restrict__ x,  const float* __restrict__ qr,
                         const float* __restrict__ cb, const float* __restrict__ sb,
                         const __bf16* __restrict__ wqb, const __bf16* __restrict__ wkb,
                         const __bf16* __restrict__ wpb,
                         const float* __restrict__ ln_w, const float* __restrict__ ln_b,
                         __bf16* __restrict__ qe, __bf16* __restrict__ kkb) {
    extern __shared__ char smem[];
    __bf16* xt    = (__bf16*)smem;                              // 16 x 2048 bf16 (64 KB)
    __bf16* qrt   = (__bf16*)(smem + 16*HH*2);                  // 16 x 512 bf16  (16 KB)
    float*  qtile = (float*)(smem + 16*HH*2 + 16*LATT*2);       // 16 x 512 f32   (32 KB)
    float*  ktile = qtile + 16*LATT;                            // 16 x 64 f32
    float*  gate  = ktile + 16*HDD;                             // 16 x 8
    float*  mu    = gate + 16*NHH;                              // 16
    float*  istd  = mu + 16;                                    // 16

    const int tid  = threadIdx.x;
    const int wave = tid >> 5, lane = tid & 31;
    const int row0 = blockIdx.x * 16;                 // global row in [0, B*S)

    // stage x, qr tiles into LDS as bf16 (f32->bf16 on the fly)
    for (int i = tid; i < 16*HH; i += 256) {
        int m = i >> 11, c = i & (HH - 1);
        xt[i] = (__bf16)x[(size_t)(row0 + m) * HH + c];
    }
    for (int i = tid; i < 16*LATT; i += 256) {
        int m = i >> 9, c = i & (LATT - 1);
        qrt[i] = (__bf16)qr[(size_t)(row0 + m) * LATT + c];
    }
    __syncthreads();

    const int mm    = lane & 15;
    const int koff  = (lane & 16) ? 16 : 0;
    const int boff  = (lane & 16) ? 16 : 0;
    const int mbase = (lane & 16) ? 8 : 0;

    // phase 1: q projection 16x512 @ (512x512)^T : 32 N-tiles over 8 waves
    for (int nt = wave; nt < LATT / 16; nt += 8) {
        const int n0 = nt * 16;
        const __bf16* wrow = wqb + (size_t)(n0 + mm) * LATT + boff;
        v8f c = {};
        #pragma unroll
        for (int k0 = 0; k0 < LATT; k0 += 32) {
            v16bf a = load_frag2(qrt + mm*LATT + k0 + koff,
                                 qrt + mm*LATT + k0 + 8 + koff);
            v16bf b = load_frag1(wrow + k0);
            c = bwmma(a, b, c);
        }
        #pragma unroll
        for (int j = 0; j < 8; ++j) qtile[(mbase + j)*LATT + n0 + mm] = c[j];
    }
    __syncthreads();

    // phase 2: waves 0-3: k projection (one 16-output N-tile each, K=2048)
    //          wave 4   : gate projection via WMMA (padded 16-col N-tile, 8 valid)
    if (wave < 5) {
        const __bf16* wrow = (wave < 4)
            ? wkb + (size_t)(wave * 16 + mm) * HH + boff
            : wpb + (size_t)mm * HH + boff;
        v8f c = {};
        for (int k0 = 0; k0 < HH; k0 += 32) {
            v16bf a = load_frag2(xt + mm*HH + k0 + koff,
                                 xt + mm*HH + k0 + 8 + koff);
            v16bf b = load_frag1(wrow + k0);
            c = bwmma(a, b, c);
        }
        if (wave < 4) {
            const int n0 = wave * 16;
            #pragma unroll
            for (int j = 0; j < 8; ++j) ktile[(mbase + j)*HDD + n0 + mm] = c[j];
        } else if (mm < NHH) {   // store-side mask only; WMMA ran with full EXEC
            #pragma unroll
            for (int j = 0; j < 8; ++j)
                gate[(mbase + j)*NHH + mm] = c[j] * 0.35355339059327373f;  // * NH^-0.5
        }
    }
    __syncthreads();

    // LN statistics per row
    if (tid < 16) {
        float s = 0.f, ss = 0.f;
        for (int d = 0; d < HDD; ++d) { float v = ktile[tid*HDD + d]; s += v; ss += v*v; }
        float m_ = s * (1.f / HDD);
        mu[tid]   = m_;
        istd[tid] = rsqrtf(ss * (1.f / HDD) - m_*m_ + 1e-5f);
    }
    __syncthreads();

    // phase 3: LN + rope + gate-fold + scale, emit bf16 qe / kk
    const int b     = row0 / SS;
    const int srow0 = row0 - b * SS;
    for (int i = tid; i < 16*HDD; i += 256) {
        int m = i >> 6, d = i & 63;
        size_t rp = ((size_t)b * SS + (srow0 + m)) * HDD + d;
        float cs = cb[rp], sn = sb[rp];
        int   p   = (d < 32) ? d + 32 : d - 32;
        float sgn = (d < 32) ? -1.f : 1.f;
        // key vector
        float kv = (ktile[m*HDD + d] - mu[m]) * istd[m] * ln_w[d] + ln_b[d];
        float kp = (ktile[m*HDD + p] - mu[m]) * istd[m] * ln_w[p] + ln_b[p];
        kkb[(size_t)(row0 + m) * HDD + d] = (__bf16)(kv * cs + sgn * kp * sn);
        // gated query vector (rope is linear -> fold heads first)
        float qv = 0.f, qp = 0.f;
        #pragma unroll
        for (int h = 0; h < NHH; ++h) {
            float g = gate[m*NHH + h];
            qv += g * qtile[m*LATT + h*HDD + d];
            qp += g * qtile[m*LATT + h*HDD + p];
        }
        qe[(size_t)(row0 + m) * HDD + d] = (__bf16)((qv * cs + sgn * qp * sn) * 0.125f);
    }
}

// ---- kernel 2: scores (WMMA, K=64) into LDS + fused radix top-256 ------
__global__ void idx_score_topk(const __bf16* __restrict__ qe,
                               const __bf16* __restrict__ kkb,
                               int* __restrict__ out) {
    extern __shared__ char smem[];
    float*    sc   = (float*)smem;                        // 16 x 2048 f32 (128 KB)
    unsigned* hist = (unsigned*)(smem + 16*SS*4);         // 8 waves x 256 bins
    unsigned* bc   = hist + 8*256;                        // 8 x {prefix, remaining, cntHi, cntEq}

    const int tid  = threadIdx.x;
    const int wave = tid >> 5, lane = tid & 31;
    const int row0 = blockIdx.x * 16;
    const int b    = row0 / SS, q0 = row0 - b * SS;

    const int mm    = lane & 15;
    const int koff  = (lane & 16) ? 16 : 0;
    const int boff  = (lane & 16) ? 16 : 0;
    const int mbase = (lane & 16) ? 8 : 0;

    // A fragments (qe tile, K=64 -> two frags), reused for all 128 key tiles
    const __bf16* qrow = qe + (size_t)(row0 + mm) * HDD;
    v16bf a0 = load_frag2(qrow + 0  + koff, qrow + 8 + koff);
    v16bf a1 = load_frag2(qrow + 32 + koff, qrow + 40 + koff);

    for (int kt = wave; kt < SS / 16; kt += 8) {
        const int key0 = kt * 16;
        const __bf16* krow = kkb + (size_t)((size_t)b*SS + key0 + mm) * HDD + boff;
        if (kt + 8 < SS / 16)   // prefetch next key tile for this wave
            __builtin_prefetch(kkb + (size_t)((size_t)b*SS + key0 + 128 + mm) * HDD, 0, 0);
        v8f c = {};
        c = bwmma(a0, load_frag1(krow),      c);
        c = bwmma(a1, load_frag1(krow + 32), c);
        #pragma unroll
        for (int j = 0; j < 8; ++j) {
            int m = mbase + j;
            float v = c[j];
            if (key0 + mm > q0 + m) v += -1.0e9f;         // causal mask
            sc[m*SS + key0 + mm] = v;
        }
    }
    __syncthreads();

    // top-256 per row: 4-pass radix select on order-preserving uint keys.
    // Every wave runs an identical barrier schedule (2 rows each).
    for (int rr = 0; rr < 2; ++rr) {
        const int m    = wave * 2 + rr;
        const float* row = sc + m * SS;
        const int obase  = (b * SS + (q0 + m)) * TOPKK;

        unsigned prefix = 0; int remaining = TOPKK;
        for (int shift = 24; shift >= 0; shift -= 8) {
            for (int i = lane; i < 256; i += 32) hist[wave*256 + i] = 0u;
            __syncthreads();
            unsigned himask = (shift == 24) ? 0u : (0xFFFFFFFFu << (shift + 8));
            for (int i = lane; i < SS; i += 32) {
                unsigned u = key_of(row[i]);
                if ((u & himask) == (prefix & himask))
                    atomicAdd(&hist[wave*256 + ((u >> shift) & 255u)], 1u);
            }
            __syncthreads();
            if (lane == 0) {
                int cum = 0; unsigned sel = 0; int rem2 = remaining;
                for (int bin = 255; bin >= 0; --bin) {
                    int cnt = (int)hist[wave*256 + bin];
                    if (cum + cnt >= remaining) { sel = (unsigned)bin; rem2 = remaining - cum; break; }
                    cum += cnt;
                }
                bc[wave*4 + 0] = prefix | (sel << shift);
                bc[wave*4 + 1] = (unsigned)rem2;
                bc[wave*4 + 2] = 0u;
                bc[wave*4 + 3] = 0u;
            }
            __syncthreads();
            prefix    = bc[wave*4 + 0];
            remaining = (int)bc[wave*4 + 1];
            __syncthreads();
        }
        const int nHi = TOPKK - remaining;     // # strictly above threshold
        for (int i = lane; i < SS; i += 32) {
            if (key_of(row[i]) > prefix) {
                int slot = (int)atomicAdd(&bc[wave*4 + 2], 1u);
                out[obase + slot] = i;
            }
        }
        for (int i = lane; i < SS; i += 32) {
            if (key_of(row[i]) == prefix) {
                int slot = (int)atomicAdd(&bc[wave*4 + 3], 1u);
                if (slot < remaining) out[obase + nHi + slot] = i;
            }
        }
        __syncthreads();
    }
}

// ---- launcher ----------------------------------------------------------
extern "C" void kernel_launch(void* const* d_in, const int* in_sizes, int n_in,
                              void* d_out, int out_size, void* d_ws, size_t ws_size,
                              hipStream_t stream) {
    const float* x    = (const float*)d_in[0];
    const float* qr   = (const float*)d_in[1];
    const float* cb   = (const float*)d_in[2];
    const float* sb   = (const float*)d_in[3];
    // d_in[4] attention_mask: unused (causal mask synthesized -> saves 33.5 MB of HBM reads)
    const float* wq_b = (const float*)d_in[5];
    const float* wk_w = (const float*)d_in[6];
    const float* ln_w = (const float*)d_in[7];
    const float* ln_b = (const float*)d_in[8];
    const float* wp_w = (const float*)d_in[9];

    char* ws = (char*)d_ws;
    size_t off = 0;
    __bf16* wqb = (__bf16*)(ws + off); off += (size_t)LATT*NHH*HDD*2;   // 512 KB
    __bf16* wkb = (__bf16*)(ws + off); off += (size_t)HDD*HH*2;         // 256 KB
    __bf16* wpb = (__bf16*)(ws + off); off += (size_t)16*HH*2;          //  64 KB (padded)
    __bf16* qe  = (__bf16*)(ws + off); off += (size_t)BB*SS*HDD*2;      // 512 KB
    __bf16* kkb = (__bf16*)(ws + off);                                   // 512 KB
    int*    out = (int*)d_out;

    idx_convert_w<<<(LATT*NHH*HDD + 255)/256, 256, 0, stream>>>(wq_b, wk_w, wp_w,
                                                                wqb, wkb, wpb);

    size_t lds1 = (size_t)16*HH*2 + 16*LATT*2
                + ((size_t)16*LATT + 16*HDD + 16*NHH + 32) * 4;
    idx_prep<<<BB*SS/16, 256, lds1, stream>>>(x, qr, cb, sb, wqb, wkb, wpb,
                                              ln_w, ln_b, qe, kkb);

    size_t lds2 = (size_t)16*SS*4 + 8*256*4 + 8*4*4;
    idx_score_topk<<<BB*SS/16, 256, lds2, stream>>>(qe, kkb, out);
}
